// _SelfAttLogn_50302656971161
// MI455X (gfx1250) — matrix operand order repaired and checked by reading, not compiled, not run on hardware
//
#include <hip/hip_runtime.h>
#include <hip/hip_bf16.h>
#include <stdint.h>

// ---------------------------------------------------------------------------
// LogN self-attention for MI455X (gfx1250, wave32, WMMA)
//   B=8, T=2048, D_MODEL=1024, H=16, DH=64, LOG_LEN=11
// Dominated by 4 GEMMs (16384x1024x1024) -> bf16 WMMA w/ f32 accumulate.
// ---------------------------------------------------------------------------

#define B_   8
#define T_   2048
#define DM   1024
#define H_   16
#define DH_  64
#define M_   (B_ * T_)                 // 16384 rows
#define MD   ((size_t)M_ * DM)         // 16777216 elements per [M, 1024] buffer
#define NEG9 (-1.0e9f)

typedef __attribute__((ext_vector_type(16))) __bf16 v16bf;
typedef __attribute__((ext_vector_type(2)))  __bf16 v2bf;
typedef __attribute__((ext_vector_type(8)))  float  v8f;

union AFrag { v16bf v; uint32_t u[8]; };
union CFrag { v8f   v; float    f[8]; };
union P2    { v2bf  v; uint32_t u; uint16_t h[2]; };

// Packed f32 -> bf16 (RNE): native casts -> backend emits v_cvt_pk_bf16_f32.
__device__ __forceinline__ P2 pack2bf(float a, float b) {
    P2 p;
    p.v = v2bf{(__bf16)a, (__bf16)b};
    return p;
}

// ---------------------------------------------------------------------------
// GEMM: C[M,N] = A[M,K] * Bw[K,N] * scale    (f32 in/out, bf16 WMMA inside)
// Block tile 128x128, K-step 32. 256 threads = 8 waves; wave -> 32x64 subtile
// (2x4 v_wmma_f32_16x16x32_bf16 tiles, f32 accumulators).
// ---------------------------------------------------------------------------
#define BM 128
#define BN 128
#define BK 32
#define LDS_STR 34   // uint16 stride: 32 + 2 pad (4B aligned pairs, bank-spread)

__global__ __launch_bounds__(256) void gemm_bf16_wmma(
    const float* __restrict__ A, const float* __restrict__ Bw,
    float* __restrict__ C, int Msz, int Nsz, int Ksz, float scale)
{
    __shared__ uint16_t As[BM][LDS_STR];   // A tile, row-major [m][k]
    __shared__ uint16_t Bs[BN][LDS_STR];   // B tile, transposed [n][k] (K-pairs contiguous)

    const int tid   = threadIdx.x;
    const int lane  = tid & 31;
    const int wave  = tid >> 5;
    const int wm    = wave & 3;            // 4 waves down   (32 rows each)
    const int wn    = wave >> 2;           // 2 waves across (64 cols each)
    const int mbase = blockIdx.y * BM;
    const int nbase = blockIdx.x * BN;
    const int r16   = lane & 15;
    const int rhalf = lane >> 4;

    CFrag acc[2][4];
    #pragma unroll
    for (int i = 0; i < 2; ++i)
        #pragma unroll
        for (int j = 0; j < 4; ++j)
            #pragma unroll
            for (int r = 0; r < 8; ++r) acc[i][j].f[r] = 0.0f;

    for (int k0 = 0; k0 < Ksz; k0 += BK) {
        // ---- stage A tile: 128x32 f32 -> bf16 LDS (4 float4 per thread) ----
        #pragma unroll
        for (int i = 0; i < 4; ++i) {
            int idx = tid + 256 * i;               // 0..1023 float4 slots
            int row = idx >> 3;
            int c4  = (idx & 7) << 2;
            const float4 f = *reinterpret_cast<const float4*>(
                A + (size_t)(mbase + row) * Ksz + k0 + c4);
            *reinterpret_cast<uint32_t*>(&As[row][c4])     = pack2bf(f.x, f.y).u;
            *reinterpret_cast<uint32_t*>(&As[row][c4 + 2]) = pack2bf(f.z, f.w).u;
        }
        // ---- stage B tile: 32x128 f32, transposed into Bs[n][k] ----
        #pragma unroll
        for (int i = 0; i < 4; ++i) {
            int idx = tid + 256 * i;               // 0..1023 float4 slots
            int kr  = idx >> 5;                    // 0..31
            int c4  = (idx & 31) << 2;             // 0..124
            const float4 f = *reinterpret_cast<const float4*>(
                Bw + (size_t)(k0 + kr) * Nsz + nbase + c4);
            const P2 p01 = pack2bf(f.x, f.y);
            const P2 p23 = pack2bf(f.z, f.w);
            Bs[c4 + 0][kr] = p01.h[0];
            Bs[c4 + 1][kr] = p01.h[1];
            Bs[c4 + 2][kr] = p23.h[0];
            Bs[c4 + 3][kr] = p23.h[1];
        }
        // L2 prefetch for next K-tile (global_prefetch_b8)
        if (k0 + BK < Ksz) {
            int pr = tid & 127;
            if (tid < 128)
                __builtin_prefetch(A + (size_t)(mbase + pr) * Ksz + k0 + BK, 0, 3);
            else
                __builtin_prefetch(Bw + (size_t)(k0 + BK + (pr >> 2)) * Nsz
                                      + nbase + (pr & 3) * 32, 0, 3);
        }
        __syncthreads();

        // ---- per-lane fragments per ISA 7.12.2 16-bit layouts ----
        AFrag a[2], b[4];
        #pragma unroll
        for (int tm = 0; tm < 2; ++tm) {
            int row = wm * 32 + tm * 16 + r16;     // A: lane%16 = M, lane/16 picks K half
            #pragma unroll
            for (int j = 0; j < 4; ++j) {
                a[tm].u[j]     = *reinterpret_cast<const uint32_t*>(&As[row][rhalf * 8 + 2 * j]);
                a[tm].u[4 + j] = *reinterpret_cast<const uint32_t*>(&As[row][16 + rhalf * 8 + 2 * j]);
            }
        }
        #pragma unroll
        for (int tn = 0; tn < 4; ++tn) {
            int col = wn * 64 + tn * 16 + r16;     // B: lane%16 = N, lane/16 picks K half
            int kb  = rhalf * 8;
            #pragma unroll
            for (int j = 0; j < 4; ++j) {
                b[tn].u[j]     = *reinterpret_cast<const uint32_t*>(&Bs[col][kb + 2 * j]);
                b[tn].u[4 + j] = *reinterpret_cast<const uint32_t*>(&Bs[col][16 + kb + 2 * j]);
            }
        }
        #pragma unroll
        for (int tm = 0; tm < 2; ++tm)
            #pragma unroll
            for (int tn = 0; tn < 4; ++tn)
                acc[tm][tn].v = __builtin_amdgcn_wmma_f32_16x16x32_bf16(
                    false, a[tm].v, false, b[tn].v,
                    (short)0, acc[tm][tn].v, false, false);
        __syncthreads();
    }

    // ---- epilogue: C/D layout -> VGPR r holds M = r + 8*(lane/16), N = lane%16 ----
    #pragma unroll
    for (int tm = 0; tm < 2; ++tm) {
        #pragma unroll
        for (int tn = 0; tn < 4; ++tn) {
            int row0 = mbase + wm * 32 + tm * 16 + rhalf * 8;
            int col  = nbase + wn * 64 + tn * 16 + r16;
            #pragma unroll
            for (int r = 0; r < 8; ++r)
                C[(size_t)(row0 + r) * Nsz + col] = acc[tm][tn].f[r] * scale;
        }
    }
}

// ---------------------------------------------------------------------------
// Span kernel (levels l=1..10): per (b, block, h):
//   sq = max_j q[j], span_k = max_j k[j], span_v = softmax(sq . k_j) @ v
// 64 threads = one per head dim.
// ---------------------------------------------------------------------------
__global__ __launch_bounds__(64) void span_kernel(
    const float* __restrict__ Q, const float* __restrict__ K,
    const float* __restrict__ V, float* __restrict__ SK,
    float* __restrict__ SV, int c, int n)
{
    __shared__ float sq[DH_];
    __shared__ float lg[1024];
    __shared__ float red[64];
    __shared__ float s_max, s_sum;

    const int bnh = blockIdx.x;
    const int h   = bnh % H_;
    const int bn  = bnh / H_;
    const int blk = bn % n;
    const int b   = bn / n;
    const int t0  = blk * c;
    const int d   = threadIdx.x;

    const size_t strd = (size_t)H_ * DH_;
    const size_t base = (((size_t)b * T_ + t0) * H_ + h) * DH_ + d;

    float mq = -3.4e38f, mk = -3.4e38f;
    for (int j = 0; j < c; ++j) {
        mq = fmaxf(mq, Q[base + (size_t)j * strd]);
        mk = fmaxf(mk, K[base + (size_t)j * strd]);
    }
    sq[d] = mq;
    SK[((size_t)bn * H_ + h) * DH_ + d] = mk;
    __syncthreads();

    for (int j = d; j < c; j += 64) {
        const float* kr = K + (((size_t)b * T_ + t0 + j) * H_ + h) * DH_;
        float s = 0.0f;
        #pragma unroll 8
        for (int dd = 0; dd < DH_; ++dd) s += sq[dd] * kr[dd];
        lg[j] = s;
    }
    __syncthreads();

    float pm = -3.4e38f;
    for (int j = d; j < c; j += 64) pm = fmaxf(pm, lg[j]);
    red[d] = pm;
    __syncthreads();
    if (d == 0) {
        float m = red[0];
        for (int i = 1; i < 64; ++i) m = fmaxf(m, red[i]);
        s_max = m;
    }
    __syncthreads();

    float ps = 0.0f;
    for (int j = d; j < c; j += 64) { float e = __expf(lg[j] - s_max); lg[j] = e; ps += e; }
    red[d] = ps;
    __syncthreads();
    if (d == 0) {
        float s = 0.0f;
        for (int i = 0; i < 64; ++i) s += red[i];
        s_sum = s;
    }
    __syncthreads();

    const float inv = 1.0f / s_sum;
    float acc = 0.0f;
    for (int j = 0; j < c; ++j) acc += lg[j] * V[base + (size_t)j * strd];
    SV[((size_t)bn * H_ + h) * DH_ + d] = acc * inv;
}

// ---------------------------------------------------------------------------
// Merge kernel: one wave (32 lanes) per (b,t,h); each lane owns 2 of DH=64 dims.
// 12 logits: self q.k plus per-level q.span_k (masked), softmax, weighted sum.
// ---------------------------------------------------------------------------
__global__ __launch_bounds__(256) void merge_kernel(
    const float* __restrict__ Q, const float* __restrict__ K,
    const float* __restrict__ V, const float* __restrict__ SKb,
    const float* __restrict__ SVb, float* __restrict__ MG)
{
    const int  lane = threadIdx.x & 31;
    const int  wv   = threadIdx.x >> 5;
    const long idx  = (long)blockIdx.x * 8 + wv;   // (b*T + t)*H + h
    const int  h    = (int)(idx % H_);
    const long bt   = idx / H_;
    const int  t    = (int)(bt % T_);
    const int  b    = (int)(bt / T_);

    const size_t qb = ((size_t)bt * H_ + h) * DH_ + lane * 2;
    const float2 q2 = *reinterpret_cast<const float2*>(Q + qb);

    auto dot = [&](const float* p) -> float {
        float2 kv = *reinterpret_cast<const float2*>(p + lane * 2);
        float  s  = q2.x * kv.x + q2.y * kv.y;
        #pragma unroll
        for (int off = 16; off > 0; off >>= 1)
            s += __shfl_xor(s, off, 32);           // wave32 butterfly
        return s;
    };

    float lg[12];
    bool  ok[12];
    int   bk[12];

    lg[0] = dot(K + ((size_t)bt * H_ + h) * DH_);  // self term (q pre-scaled)
    ok[0] = true; bk[0] = 0;

    #pragma unroll
    for (int l = 0; l <= 10; ++l) {
        const int  c     = 1 << l;
        const bool valid = (t >= c) && (((t >> l) & 1) == 1);
        const int  blk   = (t >> l) - 1;
        ok[l + 1] = valid;
        bk[l + 1] = blk;
        float s = NEG9;
        if (valid) {
            const float* kp;
            if (l == 0) {
                kp = K + (((size_t)b * T_ + (t - 1)) * H_ + h) * DH_;
            } else {
                const size_t off = (size_t)MD - (MD >> (l - 1));   // level offset
                const int    nl  = T_ >> l;
                kp = SKb + off + (((size_t)b * nl + blk) * H_ + h) * DH_;
            }
            s = dot(kp);
        }
        lg[l + 1] = s;
    }

    float mx = lg[0];
    #pragma unroll
    for (int i = 1; i < 12; ++i) mx = fmaxf(mx, lg[i]);
    float w[12], sum = 0.0f;
    #pragma unroll
    for (int i = 0; i < 12; ++i) { w[i] = __expf(lg[i] - mx); sum += w[i]; }
    const float inv = 1.0f / sum;

    const float2 v2 = *reinterpret_cast<const float2*>(V + qb);
    float m0 = w[0] * inv * v2.x;
    float m1 = w[0] * inv * v2.y;

    #pragma unroll
    for (int l = 0; l <= 10; ++l) {
        if (ok[l + 1]) {
            const float* vp;
            if (l == 0) {
                vp = V + (((size_t)b * T_ + (t - 1)) * H_ + h) * DH_;
            } else {
                const size_t off = (size_t)MD - (MD >> (l - 1));
                const int    nl  = T_ >> l;
                vp = SVb + off + (((size_t)b * nl + bk[l + 1]) * H_ + h) * DH_;
            }
            const float2 s2 = *reinterpret_cast<const float2*>(vp + lane * 2);
            const float  ww = w[l + 1] * inv;
            m0 += ww * s2.x;
            m1 += ww * s2.y;
        }
    }
    *reinterpret_cast<float2*>(MG + qb) = make_float2(m0, m1);
}

// ---------------------------------------------------------------------------
// Launch: Q/K/V GEMMs -> span levels -> merge -> output GEMM.
// Workspace (f32): [Q | K | V | MERGED | SPAN_K(levels) | SPAN_V(levels)]
// ---------------------------------------------------------------------------
extern "C" void kernel_launch(void* const* d_in, const int* in_sizes, int n_in,
                              void* d_out, int out_size, void* d_ws, size_t ws_size,
                              hipStream_t stream) {
    const float* x  = (const float*)d_in[0];
    const float* Wq = (const float*)d_in[1];
    const float* Wk = (const float*)d_in[2];
    const float* Wv = (const float*)d_in[3];
    const float* Wo = (const float*)d_in[4];
    float* out = (float*)d_out;
    float* ws  = (float*)d_ws;

    float* Q  = ws;
    float* K  = ws + MD;
    float* V  = ws + 2 * MD;
    float* MG = ws + 3 * MD;
    float* SK = ws + 4 * MD;
    float* SV = ws + 5 * MD;

    const dim3 gblk(256);
    const dim3 ggrd(DM / BN, M_ / BM);   // (8, 128)

    // Q/K/V projections (q pre-scaled by 1/sqrt(DH) = 0.125)
    gemm_bf16_wmma<<<ggrd, gblk, 0, stream>>>(x, Wq, Q, M_, DM, DM, 0.125f);
    gemm_bf16_wmma<<<ggrd, gblk, 0, stream>>>(x, Wk, K, M_, DM, DM, 1.0f);
    gemm_bf16_wmma<<<ggrd, gblk, 0, stream>>>(x, Wv, V, M_, DM, DM, 1.0f);

    // Hierarchical spans, levels 1..10 (level 0 spans are K/V themselves)
    for (int l = 1; l <= 10; ++l) {
        const int    c   = 1 << l;
        const int    n   = T_ >> l;
        const size_t off = (size_t)MD - (MD >> (l - 1));
        span_kernel<<<dim3(B_ * n * H_), dim3(64), 0, stream>>>(
            Q, K, V, SK + off, SV + off, c, n);
    }

    // 12-way softmax merge
    merge_kernel<<<dim3((B_ * T_ * H_) / 8), dim3(256), 0, stream>>>(
        Q, K, V, SK, SV, MG);

    // Output projection
    gemm_bf16_wmma<<<ggrd, gblk, 0, stream>>>(MG, Wo, out, M_, DM, DM, 1.0f);
}
